// RPNNoHeadBaseMESparse_57200374448260
// MI455X (gfx1250) — compile-verified, matrix operand order checked by reading, not run on hardware
//
#include <hip/hip_runtime.h>
#include <hip/hip_bf16.h>
#include <stdint.h>
#include <stddef.h>

#ifndef __has_builtin
#define __has_builtin(x) 0
#endif

#if __has_builtin(__builtin_amdgcn_tensor_load_to_lds)
#define USE_TDM 1
#else
#define USE_TDM 0
#endif

typedef __attribute__((ext_vector_type(16))) __bf16   v16bf;
typedef __attribute__((ext_vector_type(8)))  float    v8f;
typedef __attribute__((ext_vector_type(4)))  unsigned u32x4;
typedef __attribute__((ext_vector_type(8)))  unsigned u32x8;
typedef __attribute__((ext_vector_type(4)))  unsigned tdm_u4;
typedef __attribute__((ext_vector_type(8)))  int      tdm_i8;
typedef __attribute__((ext_vector_type(4)))  int      tdm_i4;

// ---------------------------------------------------------------- helpers
__device__ __forceinline__ unsigned short f2bf(float f) {
  unsigned u = __builtin_bit_cast(unsigned, f);
  unsigned r = u + 0x7FFFu + ((u >> 16) & 1u);   // round-to-nearest-even
  return (unsigned short)(r >> 16);
}

// Build a v16bf WMMA A/B fragment from two 16-byte LDS chunks.
// CDNA5 16-bit A/B layout: lanes 0-15 hold K={0..7,16..23},
// lanes 16-31 hold K={8..15,24..31} -> two contiguous b128 loads per lane.
__device__ __forceinline__ v16bf ld_frag(const unsigned short* p0,
                                         const unsigned short* p1) {
  u32x4 a = *(const u32x4*)p0;
  u32x4 b = *(const u32x4*)p1;
  u32x8 c = __builtin_shufflevector(a, b, 0, 1, 2, 3, 4, 5, 6, 7);
  return __builtin_bit_cast(v16bf, c);
}

// ---------------------------------------------------------------- TDM weight staging
// Copies KG rows of 8KB (8 cout-tiles x 1KB fragment blocks) from the packed
// weight array into LDS via the Tensor Data Mover (TENSORcnt); fallback to
// plain cooperative loads if the builtin is unavailable.
template<int KG>
__device__ __forceinline__ void stage_weights(const unsigned short* wp_slice,
                                              unsigned short* sW, int CT,
                                              int tid, int wave) {
  __syncthreads();   // protect previous consumers of sW (and initial sA stage)
#if USE_TDM
  if (wave == 0) {
    // Generic LDS pointer: low 32 bits are the wave-relative LDS byte address.
    unsigned ldsOff = (unsigned)(uintptr_t)sW;
    unsigned long long ga = (unsigned long long)(uintptr_t)wp_slice;
    tdm_u4 g0;
    g0[0] = 1u;                                         // count=1, user mode
    g0[1] = ldsOff;                                     // lds_addr
    g0[2] = (unsigned)ga;                               // global_addr[31:0]
    g0[3] = ((unsigned)(ga >> 32) & 0x01FFFFFFu)        // global_addr[56:32]
            | (2u << 30);                               // type = 2 ("image")
    tdm_i8 g1;
    g1[0] = (2 << 16);                                  // data_size = 4B
    g1[1] = (int)(2048u << 16);                         // tensor_dim0[15:0]
    g1[2] = (int)(((unsigned)KG & 0xFFFFu) << 16);      // tensor_dim1[15:0]
    g1[3] = (int)(2048u << 16);                         // tile_dim0 = 2048 x 4B
    g1[4] = KG;                                         // tile_dim1 = KG rows
    g1[5] = CT * 256;                                   // tensor_dim0_stride (4B units)
    g1[6] = 0;
    g1[7] = 0;
    tdm_i4 gz = {0, 0, 0, 0};
#if __clang_major__ >= 23
    tdm_i8 gz8 = {0, 0, 0, 0, 0, 0, 0, 0};
    __builtin_amdgcn_tensor_load_to_lds(g0, g1, gz, gz, gz8, 0);
#else
    __builtin_amdgcn_tensor_load_to_lds(g0, g1, gz, gz, 0);
#endif
#if __has_builtin(__builtin_amdgcn_s_wait_tensorcnt)
    __builtin_amdgcn_s_wait_tensorcnt(0);
#endif
  }
#else
  for (int idx = tid; idx < KG * 512; idx += 256) {
    int r = idx >> 9, c = idx & 511;
    *(u32x4*)&sW[r * 4096 + c * 8] =
        *(const u32x4*)(wp_slice + (size_t)r * CT * 512 + c * 8);
  }
#endif
  __syncthreads();
}

// ---------------------------------------------------------------- layout passes
__global__ void __launch_bounds__(256)
nchw_to_nhwc_bf16(const float* __restrict__ x, unsigned short* __restrict__ y,
                  int C, int H, int W, int nPix) {
  int idx = blockIdx.x * 256 + threadIdx.x;
  if (idx >= nPix) return;
  int hw = H * W;
  int b = idx / hw, r = idx % hw;
  const float* src = x + (size_t)b * C * hw + r;
  unsigned short* dst = y + (size_t)idx * C;
  for (int c = 0; c < C; ++c) dst[c] = f2bf(src[(size_t)c * hw]);
}

// Fold BN into weights, convert to bf16, and permute into the fragment-ready
// packed layout: [tap][kchunk][cout_tile16][lane32][16 values].
__global__ void __launch_bounds__(256)
fold_weights(const float* __restrict__ w, const float* __restrict__ g,
             const float* __restrict__ bb, const float* __restrict__ mm,
             const float* __restrict__ vv, unsigned short* __restrict__ wp,
             float* __restrict__ bias, int CIN, int COUT, int K) {
  int packedN = K * K * CIN * COUT;
  int idx = blockIdx.x * 256 + threadIdx.x;
  if (idx < packedN) {
    int j = idx & 15;
    int lane = (idx >> 4) & 31;
    int CT = COUT >> 4, KC = CIN >> 5;
    int q = idx >> 9;
    int ct = q % CT;
    int r2 = q / CT;
    int kc = r2 % KC;
    int t = r2 / KC;
    int n = lane & 15;
    int klo = (lane & 16) ? 8 : 0;
    int kl = (j < 8) ? (klo + j) : (klo + 16 + j - 8);
    int co = ct * 16 + n;
    int ci = kc * 32 + kl;
    int ty = t / K, tx = t % K;
    float s = g[co] * rsqrtf(vv[co] + 1e-3f);
    float val = w[(((size_t)co * CIN + ci) * K + ty) * K + tx] * s;
    wp[idx] = f2bf(val);
  } else if (idx < packedN + COUT) {
    int co = idx - packedN;
    float s = g[co] * rsqrtf(vv[co] + 1e-3f);
    bias[co] = bb[co] - mm[co] * s;
  }
}

// ---------------------------------------------------------------- 3x3 conv + BN + ReLU
// Implicit GEMM: M = output pixels (4 rows x 16 cols per block), N = output
// channels (128 per block), K = 9*CIN in chunks of 32 (bf16 WMMA 16x16x32).
// Software pipeline over flattened (tap,kchunk): double-buffered A tile in
// LDS, next chunk global-loaded into registers while current WMMAs execute;
// one barrier per k-step. 8 waves: wave = (chgrp<<2)|pixrow.
template<int CIN, int S>
__global__ void __launch_bounds__(256)
conv3x3_bn_relu(const unsigned short* __restrict__ in,   // bf16 NHWC
                const unsigned short* __restrict__ wp,   // packed bf16
                const float* __restrict__ bias,
                unsigned short* __restrict__ out,        // bf16 NHWC
                int Hin, int Win, int Hout, int Wout, int COUT) {
  constexpr int KC = CIN / 32;
  constexpr int KCLOG = (KC == 4) ? 2 : 3;
  constexpr int KG = (KC > 4) ? 4 : KC;
  constexpr int TOTK = 9 * KC;
  __shared__ __align__(16) unsigned short sW[KG * 8 * 512];  // weights (<=32KB)
  __shared__ __align__(16) unsigned short sA[2 * 64 * 32];   // A ping-pong (8KB)

  const int tid = threadIdx.x;
  const int lane = tid & 31;
  const int wave = tid >> 5;
  const int rrow = wave & 3;
  const int chg = wave >> 2;
  const int ncg = COUT >> 7;
  const int b = blockIdx.z / ncg;
  const int cg = blockIdx.z % ncg;
  const int CT = COUT >> 4;
  const int ctbase = cg * 8;
  const int oy0 = blockIdx.y * 4;
  const int ox0 = blockIdx.x * 16;
  const int klo = (lane & 16) ? 8 : 0;
  const int mrow = (lane & 16) ? 8 : 0;

  // per-thread A-staging geometry (fixed): 64 pixels x 32 ch, 16B per thread
  const int p_st = tid >> 2, c_st = tid & 3;
  const int oyS = (oy0 + (p_st >> 4)) * S;
  const int oxS = (ox0 + (p_st & 15)) * S;
  const int p_cm = rrow * 16 + (lane & 15);   // compute-side pixel

  auto loadA = [&](int ks) -> u32x4 {
    int t = ks >> KCLOG, kc = ks & (KC - 1);
    int iy = oyS + t / 3 - 1;
    int ix = oxS + t % 3 - 1;
    u32x4 val = {0u, 0u, 0u, 0u};
    if ((unsigned)iy < (unsigned)Hin && (unsigned)ix < (unsigned)Win)
      val = *(const u32x4*)(in + (((size_t)b * Hin + iy) * Win + ix) * CIN +
                            kc * 32 + c_st * 8);
    return val;
  };

  v8f acc[4] = {};
  u32x4 val = loadA(0);

  for (int ks = 0; ks < TOTK; ++ks) {
    if ((ks & (KG - 1)) == 0) {   // uniform: stage next KG k-chunks of weights
      int t = ks >> KCLOG, kc0 = ks & (KC - 1);
      stage_weights<KG>(wp + ((size_t)(t * KC + kc0) * CT + ctbase) * 512,
                        sW, CT, tid, wave);
    }
    const int bsel = (ks & 1) * 2048;
    *(u32x4*)&sA[bsel + p_st * 32 + c_st * 8] = val;
    __syncthreads();
    if (ks + 1 < TOTK) val = loadA(ks + 1);   // hide global latency behind WMMA

    v16bf afrag = ld_frag(&sA[bsel + p_cm * 32 + klo],
                          &sA[bsel + p_cm * 32 + klo + 16]);
    const int kq = ks & (KG - 1);
    v16bf bfr[4];
#pragma unroll
    for (int i = 0; i < 4; ++i) {   // batch all 8 ds_load_b128, single drain
      const unsigned short* wb = &sW[(kq * 8 + chg * 4 + i) * 512 + lane * 16];
      bfr[i] = ld_frag(wb, wb + 8);
    }
#pragma unroll
    for (int i = 0; i < 4; ++i)     // back-to-back WMMA burst
      acc[i] = __builtin_amdgcn_wmma_f32_16x16x32_bf16(
          false, afrag, false, bfr[i], (short)0, acc[i], false, false);
  }

  // epilogue: bias + ReLU + bf16 NHWC store (C/D layout: M=v+mrow, N=lane&15)
  const int oy = oy0 + rrow;
#pragma unroll
  for (int i = 0; i < 4; ++i) {
    const int co = cg * 128 + chg * 64 + i * 16 + (lane & 15);
    const float bi = bias[co];
#pragma unroll
    for (int v = 0; v < 8; ++v) {
      const int ox = ox0 + mrow + v;
      if (oy < Hout && ox < Wout) {
        float r = acc[i][v] + bi;
        r = r > 0.f ? r : 0.f;
        out[(((size_t)b * Hout + oy) * Wout + ox) * COUT + co] = f2bf(r);
      }
    }
  }
}

// ---------------------------------------------------------------- deblock (k==s)
// Each of the S*S taps is an independent 1x1 GEMM over CIN; the input tile is
// staged to LDS once and reused by all taps. Output scattered to fp32 NCHW
// d_out with sub-pixel offsets (y = iy*S + ty, x = ix*S + tx). COUT = 256.
template<int CIN, int S>
__global__ void __launch_bounds__(256)
deconv_bn_relu(const unsigned short* __restrict__ in,   // bf16 NHWC
               const unsigned short* __restrict__ wp,
               const float* __restrict__ bias,
               float* __restrict__ out,                 // fp32 NCHW [B,768,HO,WO]
               int Hin, int Win, int HO, int WO, int coOff) {
  constexpr int KC = CIN / 32;
  constexpr int KG = (KC > 4) ? 4 : KC;
  constexpr int CT = 16;  // COUT=256
  __shared__ __align__(16) unsigned short sW[KG * 8 * 512];
  __shared__ __align__(16) unsigned short sA[64 * CIN];

  const int tid = threadIdx.x, lane = tid & 31, wave = tid >> 5;
  const int rrow = wave & 3, chg = wave >> 2;
  const int b = blockIdx.z >> 1, cg = blockIdx.z & 1;
  const int ctbase = cg * 8;
  const int iy0 = blockIdx.y * 4, ix0 = blockIdx.x * 16;
  const int klo = (lane & 16) ? 8 : 0;
  const int mrow = (lane & 16) ? 8 : 0;

  // stage full activation tile (64 pixels x CIN) once
  for (int idx = tid; idx < 8 * CIN; idx += 256) {
    int p = idx / (CIN / 8), c = idx % (CIN / 8);
    int iy = iy0 + (p >> 4), ix = ix0 + (p & 15);
    u32x4 val = {0u, 0u, 0u, 0u};
    if (iy < Hin && ix < Win)
      val = *(const u32x4*)(in + (((size_t)b * Hin + iy) * Win + ix) * CIN + c * 8);
    *(u32x4*)&sA[p * CIN + c * 8] = val;
  }
  // (first stage_weights barrier covers this staging)

  const int p = rrow * 16 + (lane & 15);
  const int iy = iy0 + rrow;

  for (int t = 0; t < S * S; ++t) {
    v8f zero = {};
    v8f acc[4] = {zero, zero, zero, zero};
    for (int kg = 0; kg < KC; kg += KG) {
      stage_weights<KG>(wp + ((size_t)(t * KC + kg) * CT + ctbase) * 512,
                        sW, CT, tid, wave);
#pragma unroll
      for (int kq = 0; kq < KG; ++kq) {
        int kc = kg + kq;
        v16bf afrag = ld_frag(&sA[p * CIN + kc * 32 + klo],
                              &sA[p * CIN + kc * 32 + klo + 16]);
        v16bf bfr[4];
#pragma unroll
        for (int i = 0; i < 4; ++i) {
          const unsigned short* wb = &sW[(kq * 8 + chg * 4 + i) * 512 + lane * 16];
          bfr[i] = ld_frag(wb, wb + 8);
        }
#pragma unroll
        for (int i = 0; i < 4; ++i)
          acc[i] = __builtin_amdgcn_wmma_f32_16x16x32_bf16(
              false, afrag, false, bfr[i], (short)0, acc[i], false, false);
      }
    }
    const int ty = t / S, tx = t % S;
#pragma unroll
    for (int i = 0; i < 4; ++i) {
      const int cloc = cg * 128 + chg * 64 + i * 16 + (lane & 15);
      const float bi = bias[cloc];
      const int co = coOff + cloc;
#pragma unroll
      for (int v = 0; v < 8; ++v) {
        const int ix = ix0 + mrow + v;
        if (iy < Hin && ix < Win) {
          float r = acc[i][v] + bi;
          r = r > 0.f ? r : 0.f;
          out[(((size_t)b * 768 + co) * HO + iy * S + ty) * WO + ix * S + tx] = r;
        }
      }
    }
  }
}

// ---------------------------------------------------------------- host orchestration
extern "C" void kernel_launch(void* const* d_in, const int* in_sizes, int n_in,
                              void* d_out, int out_size, void* d_ws, size_t ws_size,
                              hipStream_t stream) {
  (void)in_sizes; (void)n_in; (void)out_size; (void)ws_size;
  const float* x = (const float*)d_in[0];
  auto wP = [&](int L) { return (const float*)d_in[1 + 5 * L + 0]; };
  auto gP = [&](int L) { return (const float*)d_in[1 + 5 * L + 1]; };
  auto bP = [&](int L) { return (const float*)d_in[1 + 5 * L + 2]; };
  auto mP = [&](int L) { return (const float*)d_in[1 + 5 * L + 3]; };
  auto vP = [&](int L) { return (const float*)d_in[1 + 5 * L + 4]; };

  // Layers 0-3: block0, 4-9: block1, 10-15: block2, 16-18: deblocks.
  static const int LCIN[19]  = {128, 128, 128, 128, 128, 128, 128, 128, 128, 128,
                                128, 256, 256, 256, 256, 256, 128, 128, 256};
  static const int LCOUT[19] = {128, 128, 128, 128, 128, 128, 128, 128, 128, 128,
                                256, 256, 256, 256, 256, 256, 256, 256, 256};
  static const int LK[19]    = {3, 3, 3, 3, 3, 3, 3, 3, 3, 3,
                                3, 3, 3, 3, 3, 3, 1, 2, 4};

  char* ws = (char*)d_ws;
  size_t off = 0;
  auto alloc = [&](size_t bytes) -> void* {
    void* pt = ws + off;
    off = (off + bytes + 255) & ~(size_t)255;
    return pt;
  };
  const size_t xcElems  = (size_t)2 * 496 * 432 * 128;  // 109.7 MB bf16
  const size_t bufElems = (size_t)2 * 248 * 216 * 128;  // 27.4 MB bf16 each
  unsigned short* xc   = (unsigned short*)alloc(xcElems * 2);
  unsigned short* bufA = (unsigned short*)alloc(bufElems * 2);
  unsigned short* bufB = (unsigned short*)alloc(bufElems * 2);

  size_t wtot = 0, btot = 0;
  for (int L = 0; L < 19; ++L) {
    wtot += (size_t)LK[L] * LK[L] * LCIN[L] * LCOUT[L];
    btot += (size_t)LCOUT[L];
  }
  unsigned short* wpool = (unsigned short*)alloc(wtot * 2);
  float* bpool = (float*)alloc(btot * 4);
  unsigned short* wp[19];
  float* bp[19];
  {
    size_t wo = 0, bo = 0;
    for (int L = 0; L < 19; ++L) {
      wp[L] = wpool + wo;
      bp[L] = bpool + bo;
      wo += (size_t)LK[L] * LK[L] * LCIN[L] * LCOUT[L];
      bo += (size_t)LCOUT[L];
    }
  }

  float* outF = (float*)d_out;
  dim3 blk(256);

  // 1) fp32 NCHW -> bf16 NHWC input
  {
    int nPix = 2 * 496 * 432;
    nchw_to_nhwc_bf16<<<dim3((nPix + 255) / 256), blk, 0, stream>>>(
        x, xc, 128, 496, 432, nPix);
  }
  // 2) fold BN into packed bf16 weights + fp32 bias
  for (int L = 0; L < 19; ++L) {
    int pn = LK[L] * LK[L] * LCIN[L] * LCOUT[L];
    fold_weights<<<dim3((pn + LCOUT[L] + 255) / 256), blk, 0, stream>>>(
        wP(L), gP(L), bP(L), mP(L), vP(L), wp[L], bp[L], LCIN[L], LCOUT[L], LK[L]);
  }

  // 3) stage 1 @ 248x216 (grid: 216/16=14, 248/4=62)
  conv3x3_bn_relu<128, 2><<<dim3(14, 62, 2), blk, 0, stream>>>(xc,   wp[0], bp[0], bufA, 496, 432, 248, 216, 128);
  conv3x3_bn_relu<128, 1><<<dim3(14, 62, 2), blk, 0, stream>>>(bufA, wp[1], bp[1], bufB, 248, 216, 248, 216, 128);
  conv3x3_bn_relu<128, 1><<<dim3(14, 62, 2), blk, 0, stream>>>(bufB, wp[2], bp[2], bufA, 248, 216, 248, 216, 128);
  conv3x3_bn_relu<128, 1><<<dim3(14, 62, 2), blk, 0, stream>>>(bufA, wp[3], bp[3], bufB, 248, 216, 248, 216, 128);
  deconv_bn_relu<128, 1><<<dim3(14, 62, 4), blk, 0, stream>>>(bufB, wp[16], bp[16], outF, 248, 216, 248, 216, 0);

  // 4) stage 2 @ 124x108 (grid: 108/16=7, 124/4=31)
  conv3x3_bn_relu<128, 2><<<dim3(7, 31, 2), blk, 0, stream>>>(bufB, wp[4], bp[4], bufA, 248, 216, 124, 108, 128);
  conv3x3_bn_relu<128, 1><<<dim3(7, 31, 2), blk, 0, stream>>>(bufA, wp[5], bp[5], bufB, 124, 108, 124, 108, 128);
  conv3x3_bn_relu<128, 1><<<dim3(7, 31, 2), blk, 0, stream>>>(bufB, wp[6], bp[6], bufA, 124, 108, 124, 108, 128);
  conv3x3_bn_relu<128, 1><<<dim3(7, 31, 2), blk, 0, stream>>>(bufA, wp[7], bp[7], bufB, 124, 108, 124, 108, 128);
  conv3x3_bn_relu<128, 1><<<dim3(7, 31, 2), blk, 0, stream>>>(bufB, wp[8], bp[8], bufA, 124, 108, 124, 108, 128);
  conv3x3_bn_relu<128, 1><<<dim3(7, 31, 2), blk, 0, stream>>>(bufA, wp[9], bp[9], bufB, 124, 108, 124, 108, 128);
  deconv_bn_relu<128, 2><<<dim3(7, 31, 4), blk, 0, stream>>>(bufB, wp[17], bp[17], outF, 124, 108, 248, 216, 256);

  // 5) stage 3 @ 62x54, 256 ch (grid: 54/16->4, 62/4->16)
  conv3x3_bn_relu<128, 2><<<dim3(4, 16, 4), blk, 0, stream>>>(bufB, wp[10], bp[10], bufA, 124, 108, 62, 54, 256);
  conv3x3_bn_relu<256, 1><<<dim3(4, 16, 4), blk, 0, stream>>>(bufA, wp[11], bp[11], bufB, 62, 54, 62, 54, 256);
  conv3x3_bn_relu<256, 1><<<dim3(4, 16, 4), blk, 0, stream>>>(bufB, wp[12], bp[12], bufA, 62, 54, 62, 54, 256);
  conv3x3_bn_relu<256, 1><<<dim3(4, 16, 4), blk, 0, stream>>>(bufA, wp[13], bp[13], bufB, 62, 54, 62, 54, 256);
  conv3x3_bn_relu<256, 1><<<dim3(4, 16, 4), blk, 0, stream>>>(bufB, wp[14], bp[14], bufA, 62, 54, 62, 54, 256);
  conv3x3_bn_relu<256, 1><<<dim3(4, 16, 4), blk, 0, stream>>>(bufA, wp[15], bp[15], bufB, 62, 54, 62, 54, 256);
  deconv_bn_relu<256, 4><<<dim3(4, 16, 4), blk, 0, stream>>>(bufB, wp[18], bp[18], outF, 62, 54, 248, 216, 512);
}
